// LiMoEFusion_79551384257130
// MI455X (gfx1250) — matrix operand order repaired and checked by reading, not compiled, not run on hardware
//
#include <hip/hip_runtime.h>
#include <hip/hip_bf16.h>
#include <cstdint>

// ---------------------------------------------------------------------------
// LiMoE fusion forward for MI455X (gfx1250).
// All GEMMs run on the WMMA matrix path: f16 A/B (weights pre-transposed to
// [N][K] f16 in workspace), f32 accumulate via v_wmma_f32_16x16x32_f16.
// Tile: 64x128 per workgroup, 8 waves, each wave owns a 16x64 strip
// (4 accumulators -> 4 WMMAs per A-fragment, 2 barriers per K-step of 32).
// ---------------------------------------------------------------------------

typedef _Float16 half_t;
typedef __attribute__((ext_vector_type(16))) _Float16 v16h;
typedef __attribute__((ext_vector_type(8)))  float    v8f;

#define BM 64
#define BN 128
#define BKK 32
#define LDSTR 48   // halfs per LDS tile row (96B: 16B-aligned, staggered banks)

// --------------------------- weight convert ---------------------------------
// src fp32 [K][N] row-major  ->  dst f16 [N][K] row-major (transposed)
__global__ void conv_transpose_f16(const float* __restrict__ S,
                                   half_t* __restrict__ D_, int K, int N) {
  long i = (long)blockIdx.x * 256 + threadIdx.x;
  if (i >= (long)K * N) return;
  int k = (int)(i / N), n = (int)(i % N);
  D_[(size_t)n * K + k] = (half_t)S[i];
}

// --------------------------- layernorm -> f16 -------------------------------
__global__ __launch_bounds__(256) void ln_to_f16_kernel(
    const float* __restrict__ X, const float* __restrict__ g,
    const float* __restrict__ bta, half_t* __restrict__ Y, int D) {
  int row = blockIdx.x;
  const float* x = X + (size_t)row * D;
  __shared__ float red[2][8];
  float s = 0.f, s2 = 0.f;
  for (int c = threadIdx.x; c < D; c += 256) { float v = x[c]; s += v; s2 += v * v; }
  for (int o = 16; o >= 1; o >>= 1) { s += __shfl_xor(s, o, 32); s2 += __shfl_xor(s2, o, 32); }
  int wave = threadIdx.x >> 5, lane = threadIdx.x & 31;
  if (lane == 0) { red[0][wave] = s; red[1][wave] = s2; }
  __syncthreads();
  if (wave == 0) {
    float a = (lane < 8) ? red[0][lane] : 0.f;
    float b2 = (lane < 8) ? red[1][lane] : 0.f;
    for (int o = 4; o >= 1; o >>= 1) { a += __shfl_xor(a, o, 32); b2 += __shfl_xor(b2, o, 32); }
    if (lane == 0) { red[0][0] = a; red[1][0] = b2; }
  }
  __syncthreads();
  float mean = red[0][0] / D;
  float var  = red[1][0] / D - mean * mean;
  float rstd = rsqrtf(var + 1e-5f);
  half_t* y = Y + (size_t)row * D;
  for (int c = threadIdx.x; c < D; c += 256)
    y[c] = (half_t)((x[c] - mean) * rstd * g[c] + bta[c]);
}

// --------------------------- WMMA GEMM --------------------------------------
// C[M,N] = epilogue( A[M,K](f16) x Bt[N,K](f16) )
// epilogue: +bias[n]; act==1 -> exact GELU; *mulbuf[r,n]; *rowscale[r*rs_stride];
//           +residual[(res_mod? r%res_mod : r), n];
//           out row = rdiv ? (r/rdiv)*rmul + roff + r%rdiv : r; store f32/f16.
__global__ __launch_bounds__(256) void gemm_f16_wmma(
    const half_t* __restrict__ A, const half_t* __restrict__ Bt,
    const float* __restrict__ bias, int act,
    const half_t* __restrict__ mulbuf,
    const float* __restrict__ rowscale, int rs_stride,
    const float* __restrict__ residual, int res_mod,
    float* __restrict__ Cf32, half_t* __restrict__ Cf16,
    int M, int N, int K, int rdiv, int rmul, int roff) {
  __shared__ __align__(16) half_t As[BM * LDSTR];
  __shared__ __align__(16) half_t Bs[BN * LDSTR];

  int tid  = threadIdx.x;
  int wave = tid >> 5, lane = tid & 31;
  int m0 = (wave & 3) * 16;         // wave's M offset in tile
  int n0 = (wave >> 2) * 64;        // wave's N offset in tile (four 16-wide subtiles)
  int tileM = blockIdx.x * BM;
  int tileN = blockIdx.y * BN;

  // cooperative tile-load indices: rows of 32 halfs, 16B chunks per thread
  int lr = tid >> 2;                 // 0..63
  int lc = (tid & 3) * 8;            // 0,8,16,24
  int ar = tileM + lr; if (ar >= M) ar = M - 1;       // clamp (no divergence)
  const half_t* Arow  = A  + (size_t)ar * K + lc;
  const half_t* Brow0 = Bt + (size_t)(tileN + lr) * K + lc;
  const half_t* Brow1 = Bt + (size_t)(tileN + 64 + lr) * K + lc;

  v8f accs[4] = {};

  int lm = lane & 15, kh = lane >> 4;
  const half_t* aPtr = As + (m0 + lm) * LDSTR;
  const half_t* bPtr = Bs + (n0 + lm) * LDSTR;

  for (int k0 = 0; k0 < K; k0 += BKK) {
    uint4 av  = *(const uint4*)(Arow  + k0);
    uint4 bv0 = *(const uint4*)(Brow0 + k0);
    uint4 bv1 = *(const uint4*)(Brow1 + k0);
    if (k0 + BKK < K) {
      __builtin_prefetch(Arow  + k0 + BKK, 0, 1);
      __builtin_prefetch(Brow0 + k0 + BKK, 0, 1);
      __builtin_prefetch(Brow1 + k0 + BKK, 0, 1);
    }
    __syncthreads();
    *(uint4*)(As + lr * LDSTR + lc) = av;
    *(uint4*)(Bs + lr * LDSTR + lc) = bv0;
    *(uint4*)(Bs + (64 + lr) * LDSTR + lc) = bv1;
    __syncthreads();

    // A fragment: lanes0-15 K 0..7 & 16..23 ; lanes16-31 K 8..15 & 24..31
    v16h afrag;
    *((uint4*)&afrag)     = *(const uint4*)(aPtr + kh * 8);
    *((uint4*)&afrag + 1) = *(const uint4*)(aPtr + 16 + kh * 8);
#pragma unroll
    for (int sub = 0; sub < 4; ++sub) {
      // B fragment: lanes0-15 K 0..7 & 16..23 ; lanes16-31 K 8..15 & 24..31 (per N column)
      v16h bf;
      const half_t* bp = bPtr + sub * 16 * LDSTR;
      *((uint4*)&bf)     = *(const uint4*)(bp + kh * 8);
      *((uint4*)&bf + 1) = *(const uint4*)(bp + 16 + kh * 8);
      accs[sub] = __builtin_amdgcn_wmma_f32_16x16x32_f16(false, afrag, false, bf,
                                                         (short)0, accs[sub], false, false);
    }
  }

  // epilogue: C/D layout — lane%16 = column, VGPR r -> row r + 8*(lane/16)
  int lg = lane >> 4;
#pragma unroll
  for (int sub = 0; sub < 4; ++sub) {
    int ncol = tileN + n0 + sub * 16 + lm;
    float bv = bias ? bias[ncol] : 0.f;
#pragma unroll
    for (int r = 0; r < 8; ++r) {
      int row = tileM + m0 + lg * 8 + r;
      if (row >= M) continue;
      float v = accs[sub][r] + bv;
      if (act == 1) v = 0.5f * v * (1.f + erff(v * 0.70710678118654752f));
      if (mulbuf)   v *= (float)mulbuf[(size_t)row * N + ncol];
      if (rowscale) v *= rowscale[(size_t)row * rs_stride];
      if (residual) {
        int rr = res_mod ? (row % res_mod) : row;
        v += residual[(size_t)rr * N + ncol];
      }
      int orow = rdiv ? ((row / rdiv) * rmul + roff + (row % rdiv)) : row;
      if (Cf32) Cf32[(size_t)orow * N + ncol] = v;
      if (Cf16) Cf16[(size_t)orow * N + ncol] = (half_t)v;
    }
  }
}

// --------------------------- attention --------------------------------------
// Q: f32 [qrows][512] (qbcast: row = n, else b*N+n). KV: f32 [B*Mk][1024]
// (k at [0:512], v at [512:1024], head h at h*128). O: f16 [B*N][512].
__global__ __launch_bounds__(256) void attn_kernel(
    const float* __restrict__ Q, const float* __restrict__ KV,
    half_t* __restrict__ O, int N, int Mk, int qbcast) {
  extern __shared__ float sm[];
  float* sQ = sm;                 // N*128
  float* sK = sQ + N * 128;       // Mk*128
  float* sS = sK + Mk * 128;      // N*Mk
  int b = blockIdx.x, h = blockIdx.y;
  int tid = threadIdx.x;

  for (int i = tid; i < N * 128; i += 256) {
    int n = i >> 7, dd = i & 127;
    int qr = qbcast ? n : b * N + n;
    sQ[i] = Q[(size_t)qr * 512 + h * 128 + dd];
  }
  for (int i = tid; i < Mk * 128; i += 256) {
    int m = i >> 7, dd = i & 127;
    sK[i] = KV[((size_t)b * Mk + m) * 1024 + h * 128 + dd];
  }
  __syncthreads();

  const float scale = 0.08838834764831845f;  // 128^-0.5
  for (int i = tid; i < N * Mk; i += 256) {
    int n = i / Mk, m = i % Mk;
    const float* qr = sQ + n * 128;
    const float* kr = sK + m * 128;
    float s = 0.f;
#pragma unroll 8
    for (int dd = 0; dd < 128; ++dd) s += qr[dd] * kr[dd];
    sS[i] = s * scale;
  }
  __syncthreads();

  int wave = tid >> 5, lane = tid & 31;
  for (int n = wave; n < N; n += 8) {
    float* row = sS + n * Mk;
    float mx = -1e30f;
    for (int m = lane; m < Mk; m += 32) mx = fmaxf(mx, row[m]);
    for (int o = 16; o >= 1; o >>= 1) mx = fmaxf(mx, __shfl_xor(mx, o, 32));
    float sum = 0.f;
    for (int m = lane; m < Mk; m += 32) { float e = __expf(row[m] - mx); row[m] = e; sum += e; }
    for (int o = 16; o >= 1; o >>= 1) sum += __shfl_xor(sum, o, 32);
    float inv = 1.f / sum;
    for (int m = lane; m < Mk; m += 32) row[m] *= inv;
  }
  __syncthreads();

  for (int i = tid; i < N * 128; i += 256) {
    int n = i >> 7, dd = i & 127;
    const float* srow = sS + n * Mk;
    const float* vcol = KV + (size_t)b * Mk * 1024 + 512 + h * 128 + dd;
    float o = 0.f;
    for (int m = 0; m < Mk; ++m) o += srow[m] * vcol[(size_t)m * 1024];
    O[((size_t)b * N + n) * 512 + h * 128 + dd] = (half_t)o;
  }
}

// --------------------------- MoE gating -------------------------------------
// logits = (h + mfeat[mod]) @ gate[512,4]; softmax -> top2 -> renorm -> cw[T,4]
__global__ __launch_bounds__(256) void gate_kernel(
    const half_t* __restrict__ H, const float* __restrict__ gate,
    const float* __restrict__ mfeat, float* __restrict__ cw, int T) {
  int wave = threadIdx.x >> 5, lane = threadIdx.x & 31;
  int tok = blockIdx.x * 8 + wave;
  if (tok >= T) return;
  int mod = (tok & 31) >> 4;
  const half_t* h = H + (size_t)tok * 512;
  const float* mf = mfeat + mod * 512;
  float a0 = 0.f, a1 = 0.f, a2 = 0.f, a3 = 0.f;
  for (int k = lane; k < 512; k += 32) {
    float v = (float)h[k] + mf[k];
    const float* g4 = gate + k * 4;
    a0 += v * g4[0]; a1 += v * g4[1]; a2 += v * g4[2]; a3 += v * g4[3];
  }
  for (int o = 16; o >= 1; o >>= 1) {
    a0 += __shfl_xor(a0, o, 32); a1 += __shfl_xor(a1, o, 32);
    a2 += __shfl_xor(a2, o, 32); a3 += __shfl_xor(a3, o, 32);
  }
  if (lane == 0) {
    float l[4] = {a0, a1, a2, a3};
    float mx = fmaxf(fmaxf(l[0], l[1]), fmaxf(l[2], l[3]));
    float w[4]; float s = 0.f;
    for (int e = 0; e < 4; ++e) { w[e] = __expf(l[e] - mx); s += w[e]; }
    for (int e = 0; e < 4; ++e) w[e] /= s;
    int i0 = 0;
    for (int e = 1; e < 4; ++e) if (w[e] > w[i0]) i0 = e;
    int i1 = -1;
    for (int e = 0; e < 4; ++e) { if (e == i0) continue; if (i1 < 0 || w[e] > w[i1]) i1 = e; }
    float r = w[i0] + w[i1];
    float o4[4] = {0.f, 0.f, 0.f, 0.f};
    o4[i0] = w[i0] / r; o4[i1] = w[i1] / r;
    float* c = cw + (size_t)tok * 4;
    c[0] = o4[0]; c[1] = o4[1]; c[2] = o4[2]; c[3] = o4[3];
  }
}

// --------------------------- pool + head ------------------------------------
__global__ void pool_kernel(const float* __restrict__ X, float* __restrict__ P) {
  int i = blockIdx.x * 256 + threadIdx.x;  // B*1024
  if (i >= 512 * 1024) return;
  int b = i >> 10, col = i & 1023;
  int m = col >> 9, c = col & 511;
  const float* base = X + ((size_t)b * 32 + m * 16) * 512 + c;
  float s = 0.f;
#pragma unroll
  for (int t = 0; t < 16; ++t) s += base[(size_t)t * 512];
  P[i] = s * (1.f / 16.f);
}

__global__ void head_kernel(const float* __restrict__ P, const float* __restrict__ W,
                            const float* __restrict__ bias, float* __restrict__ out) {
  int i = blockIdx.x * 256 + threadIdx.x;  // B*20
  if (i >= 512 * 20) return;
  int b = i / 20, p = i % 20;
  const float* pr = P + (size_t)b * 1024;
  float s = bias[p];
  for (int k = 0; k < 1024; ++k) s += pr[k] * W[k * 20 + p];
  out[i] = s;
}

// ---------------------------------------------------------------------------
struct QFIdx { int qt, cn_g, cn_b, cp_w, cp_b, n1_g, n1_b, q, kv, pw, pb,
                   n2_g, n2_b, fc1_w, fc1_b, fc2_w, fc2_b; };
struct BlkIdx { int n1_g, n1_b, q, kv, pw, pb, n2_g, n2_b, gate, mfeat, wg, wu, w2; };

extern "C" void kernel_launch(void* const* d_in, const int* in_sizes, int n_in,
                              void* d_out, int out_size, void* d_ws, size_t ws_size,
                              hipStream_t stream) {
  // ---- input index tables: insertion-order vs sorted-key flattening --------
  QFIdx qf[2]; BlkIdx bk[3]; int ixa, ixb, ihw, ihb;
  bool insertion = (n_in > 0 && in_sizes[0] > 1000000);  // xa first => insertion
  if (insertion) {
    ixa = 0; ixb = 1;
    for (int m = 0; m < 2; ++m) {
      int b = 2 + 17 * m;
      qf[m] = {b+0, b+1, b+2, b+3, b+4, b+5, b+6, b+7, b+8, b+9, b+10,
               b+11, b+12, b+13, b+14, b+15, b+16};
    }
    for (int k = 0; k < 3; ++k) {
      int b = 36 + 13 * k;
      bk[k] = {b+0, b+1, b+2, b+3, b+4, b+5, b+6, b+7, b+8, b+9, b+10, b+11, b+12};
    }
    ihw = 75; ihb = 76;
  } else {
    // sorted keys: params{blocks, head_b, head_w, proj{a,b}}, xa, xb
    for (int k = 0; k < 3; ++k) {
      int b = 13 * k;  // attn{kv,pb,pw,q}, gate, mfeat, n1_b, n1_g, n2_b, n2_g, w2, wg, wu
      bk[k] = {b+7, b+6, b+3, b+0, b+2, b+1, b+9, b+8, b+4, b+5, b+11, b+12, b+10};
    }
    ihb = 39; ihw = 40;
    for (int m = 0; m < 2; ++m) {
      int b = 41 + 17 * m;  // attn{kv,pb,pw,q}, cn_b, cn_g, cp_b, cp_w, fc1_b, fc1_w,
                            // fc2_b, fc2_w, n1_b, n1_g, n2_b, n2_g, qt
      qf[m] = {b+16, b+5, b+4, b+7, b+6, b+13, b+12, b+3, b+0, b+2, b+1,
               b+15, b+14, b+9, b+8, b+11, b+10};
    }
    ixa = 75; ixb = 76;
  }
  auto F32 = [&](int i) { return (const float*)d_in[i]; };

  // ---- workspace bump allocator -------------------------------------------
  size_t off = 0;
  auto alloc = [&](size_t bytes) -> void* {
    void* p = (char*)d_ws + off;
    off = (off + bytes + 255) & ~(size_t)255;
    return p;
  };
  auto h16  = [&](size_t n) { return (half_t*)alloc(n * sizeof(half_t)); };
  auto f32a = [&](size_t n) { return (float*)alloc(n * sizeof(float)); };
  auto transposeW = [&](const float* src, int K, int N) {
    half_t* dst = h16((size_t)K * N);
    long tot = (long)K * N;
    conv_transpose_f16<<<dim3((unsigned)((tot + 255) / 256)), dim3(256), 0, stream>>>(src, dst, K, N);
    return dst;
  };

  // ---- convert all GEMM weights to transposed f16 --------------------------
  int Fm[2] = {768, 1024};
  half_t *cpT[2], *qT[2], *kvT[2], *pwT[2], *fc1T[2], *fc2T[2];
  for (int m = 0; m < 2; ++m) {
    cpT[m]  = transposeW(F32(qf[m].cp_w),  Fm[m], 512);
    qT[m]   = transposeW(F32(qf[m].q),     512, 512);
    kvT[m]  = transposeW(F32(qf[m].kv),    512, 1024);
    pwT[m]  = transposeW(F32(qf[m].pw),    512, 512);
    fc1T[m] = transposeW(F32(qf[m].fc1_w), 512, 2048);
    fc2T[m] = transposeW(F32(qf[m].fc2_w), 2048, 512);
  }
  half_t *bqT[3], *bkvT[3], *bpwT[3], *wgT[3][4], *wuT[3][4], *w2T[3][4];
  for (int k = 0; k < 3; ++k) {
    bqT[k]  = transposeW(F32(bk[k].q),  512, 512);
    bkvT[k] = transposeW(F32(bk[k].kv), 512, 1024);
    bpwT[k] = transposeW(F32(bk[k].pw), 512, 512);
    for (int e = 0; e < 4; ++e) {
      wgT[k][e] = transposeW(F32(bk[k].wg) + (size_t)e * 512 * 2048, 512, 2048);
      wuT[k][e] = transposeW(F32(bk[k].wu) + (size_t)e * 512 * 2048, 512, 2048);
      w2T[k][e] = transposeW(F32(bk[k].w2) + (size_t)e * 2048 * 512, 2048, 512);
    }
  }

  // ---- activation buffers --------------------------------------------------
  half_t* lnx   = h16((size_t)32768 * 1024);  // LN(raw input), per modality
  half_t* ctx   = h16((size_t)32768 * 512);   // qformer context (f16)
  float*  kvbuf = f32a((size_t)32768 * 1024); // kv projections
  float*  qbuf  = f32a((size_t)16384 * 512);  // q projections
  half_t* obuf  = h16((size_t)16384 * 512);   // attention output (f16)
  float*  ybuf  = f32a((size_t)8192 * 512);   // qformer residual stream
  half_t* lnq   = h16((size_t)16 * 512);      // LN(query tokens)
  half_t* lnh   = h16((size_t)16384 * 512);   // LN for MLP / MoE input
  half_t* g1    = h16((size_t)16384 * 2048);  // fc1 out / gelu gate branch
  half_t* g2    = h16((size_t)16384 * 2048);  // GLU product
  float*  xcat  = f32a((size_t)16384 * 512);  // concatenated sequence [B*32,512]
  float*  cw    = f32a((size_t)16384 * 4);    // MoE combine weights
  float*  pooled= f32a((size_t)512 * 1024);
  (void)ws_size; (void)out_size;

  auto gemm = [&](const half_t* A, const half_t* Bt, int M, int N, int K,
                  const float* bias, int act, const half_t* mulbuf,
                  const float* rowscale, int rs_stride,
                  const float* residual, int res_mod,
                  float* Cf32, half_t* Cf16, int rdiv, int rmul, int roff) {
    dim3 grid((M + BM - 1) / BM, N / BN);
    gemm_f16_wmma<<<grid, 256, 0, stream>>>(A, Bt, bias, act, mulbuf, rowscale,
                                            rs_stride, residual, res_mod,
                                            Cf32, Cf16, M, N, K, rdiv, rmul, roff);
  };

  // ======================= Q-Former (per modality) ==========================
  const float* xin[2] = {F32(ixa), F32(ixb)};
  const int Rctx = 512 * 64;   // 32768 context rows
  const int Rq   = 512 * 16;   // 8192 query rows
  for (int m = 0; m < 2; ++m) {
    int F = Fm[m];
    ln_to_f16_kernel<<<Rctx, 256, 0, stream>>>(xin[m], F32(qf[m].cn_g), F32(qf[m].cn_b), lnx, F);
    // ctx = LN(x) @ cp_w + cp_b                         -> f16 [32768,512]
    gemm(lnx, cpT[m], Rctx, 512, F, F32(qf[m].cp_b), 0, nullptr, nullptr, 0,
         nullptr, 0, nullptr, ctx, 0, 0, 0);
    // batch-independent query projection: LN(qt) @ q    -> f32 [16,512]
    ln_to_f16_kernel<<<16, 256, 0, stream>>>(F32(qf[m].qt), F32(qf[m].n1_g), F32(qf[m].n1_b), lnq, 512);
    gemm(lnq, qT[m], 16, 512, 512, nullptr, 0, nullptr, nullptr, 0,
         nullptr, 0, qbuf, nullptr, 0, 0, 0);
    // kv = ctx @ kv_w                                   -> f32 [32768,1024]
    gemm(ctx, kvT[m], Rctx, 1024, 512, nullptr, 0, nullptr, nullptr, 0,
         nullptr, 0, kvbuf, nullptr, 0, 0, 0);
    // cross attention (16 queries, 64 keys, 4 heads)    -> f16 [8192,512]
    size_t sh = (size_t)(16 * 128 + 64 * 128 + 16 * 64) * 4;
    attn_kernel<<<dim3(512, 4), 256, sh, stream>>>(qbuf, kvbuf, obuf, 16, 64, 1);
    // y = qt + attn_o @ pw + pb                         -> f32 [8192,512]
    gemm(obuf, pwT[m], Rq, 512, 512, F32(qf[m].pb), 0, nullptr, nullptr, 0,
         F32(qf[m].qt), 16, ybuf, nullptr, 0, 0, 0);
    // MLP with residual, interleaved into xcat
    ln_to_f16_kernel<<<Rq, 256, 0, stream>>>(ybuf, F32(qf[m].n2_g), F32(qf[m].n2_b), lnh, 512);
    gemm(lnh, fc1T[m], Rq, 2048, 512, F32(qf[m].fc1_b), 1, nullptr, nullptr, 0,
         nullptr, 0, nullptr, g1, 0, 0, 0);
    gemm(g1, fc2T[m], Rq, 512, 2048, F32(qf[m].fc2_b), 0, nullptr, nullptr, 0,
         ybuf, 0, xcat, nullptr, /*remap*/ 16, 32, m * 16);
  }

  // ======================= fusion blocks ====================================
  const int T = 512 * 32;  // 16384 tokens
  for (int k = 0; k < 3; ++k) {
    ln_to_f16_kernel<<<T, 256, 0, stream>>>(xcat, F32(bk[k].n1_g), F32(bk[k].n1_b), lnh, 512);
    gemm(lnh, bqT[k], T, 512, 512, nullptr, 0, nullptr, nullptr, 0,
         nullptr, 0, qbuf, nullptr, 0, 0, 0);
    gemm(lnh, bkvT[k], T, 1024, 512, nullptr, 0, nullptr, nullptr, 0,
         nullptr, 0, kvbuf, nullptr, 0, 0, 0);
    size_t sh = (size_t)(32 * 128 + 32 * 128 + 32 * 32) * 4;
    attn_kernel<<<dim3(512, 4), 256, sh, stream>>>(qbuf, kvbuf, obuf, 32, 32, 0);
    // x += attn_o @ pw + pb   (in-place residual)
    gemm(obuf, bpwT[k], T, 512, 512, F32(bk[k].pb), 0, nullptr, nullptr, 0,
         xcat, 0, xcat, nullptr, 0, 0, 0);
    // MoE
    ln_to_f16_kernel<<<T, 256, 0, stream>>>(xcat, F32(bk[k].n2_g), F32(bk[k].n2_b), lnh, 512);
    gate_kernel<<<T / 8, 256, 0, stream>>>(lnh, F32(bk[k].gate), F32(bk[k].mfeat), cw, T);
    for (int e = 0; e < 4; ++e) {
      // G = gelu(h @ wg)                                -> f16
      gemm(lnh, wgT[k][e], T, 2048, 512, nullptr, 1, nullptr, nullptr, 0,
           nullptr, 0, nullptr, g1, 0, 0, 0);
      // He = G * (h @ wu)                               -> f16
      gemm(lnh, wuT[k][e], T, 2048, 512, nullptr, 0, g1, nullptr, 0,
           nullptr, 0, nullptr, g2, 0, 0, 0);
      // x += cw[:,e] * (He @ w2)   (in-place, stream-ordered over experts)
      gemm(g2, w2T[k][e], T, 512, 2048, nullptr, 0, nullptr, cw + e, 4,
           xcat, 0, xcat, nullptr, 0, 0, 0);
    }
  }

  // ======================= pool + head ======================================
  pool_kernel<<<(512 * 1024 + 255) / 256, 256, 0, stream>>>(xcat, pooled);
  head_kernel<<<(512 * 20 + 255) / 256, 256, 0, stream>>>(pooled, F32(ihw), F32(ihb),
                                                          (float*)d_out);
}
